// Decoder_42271068127492
// MI455X (gfx1250) — matrix operand order repaired and checked by reading, not compiled, not run on hardware
//
#include <hip/hip_runtime.h>

#define NW 32000
#define E  256
#define B  64
#define M  1024
#define F  4
#define PL 128

typedef __attribute__((ext_vector_type(16))) __bf16          v16bf;
typedef __attribute__((ext_vector_type(16))) unsigned short  v16us;
typedef __attribute__((ext_vector_type(8)))  unsigned short  v8us;
typedef __attribute__((ext_vector_type(8)))  float           v8f;
typedef __attribute__((ext_vector_type(4)))  float           v4f;

__device__ __forceinline__ unsigned short f2bf(float f) {
    unsigned int u = __builtin_bit_cast(unsigned int, f);
    u += 0x7FFFu + ((u >> 16) & 1u);          // round-to-nearest-even
    return (unsigned short)(u >> 16);
}

// ---------------------------------------------------------------------------
// K0: m = embed(tables[1], y); h = GRU(m, h_prev); enc = prof @ Wp.T + bp
// grid=B, block=E
// ---------------------------------------------------------------------------
__global__ void gru_enc_kernel(const float* __restrict__ tables,
                               const float* __restrict__ Wih,
                               const float* __restrict__ Whh,
                               const float* __restrict__ bih,
                               const float* __restrict__ bhh,
                               const float* __restrict__ Wp,
                               const float* __restrict__ bp,
                               const int*   __restrict__ y,
                               const float* __restrict__ hprev,
                               const float* __restrict__ prof,
                               float* __restrict__ q,
                               float* __restrict__ enc,
                               float* __restrict__ out_h) {
    const int b = blockIdx.x, e = threadIdx.x;
    __shared__ float sm[E], sh[E], sp[PL];
    const int yi = y[b];
    const float* tab1 = tables + (size_t)NW * E;   // C[0] = tables[1]
    sm[e] = yi ? tab1[(size_t)yi * E + e] : 0.f;
    sh[e] = hprev[b * E + e];
    if (e < PL) sp[e] = prof[b * PL + e];
    __syncthreads();

    float gir = bih[e], giz = bih[E + e], gin = bih[2 * E + e];
    float ghr = bhh[e], ghz = bhh[E + e], ghn = bhh[2 * E + e];
    const float* wi0 = Wih + (size_t)e * E;
    const float* wi1 = Wih + (size_t)(E + e) * E;
    const float* wi2 = Wih + (size_t)(2 * E + e) * E;
    const float* wh0 = Whh + (size_t)e * E;
    const float* wh1 = Whh + (size_t)(E + e) * E;
    const float* wh2 = Whh + (size_t)(2 * E + e) * E;
    for (int k = 0; k < E; ++k) {
        const float mk = sm[k], hk = sh[k];
        gir = fmaf(wi0[k], mk, gir); giz = fmaf(wi1[k], mk, giz); gin = fmaf(wi2[k], mk, gin);
        ghr = fmaf(wh0[k], hk, ghr); ghz = fmaf(wh1[k], hk, ghz); ghn = fmaf(wh2[k], hk, ghn);
    }
    const float r = 1.f / (1.f + __expf(-(gir + ghr)));
    const float z = 1.f / (1.f + __expf(-(giz + ghz)));
    const float n = tanhf(gin + r * ghn);
    const float h = (1.f - z) * n + z * sh[e];
    q[b * E + e]     = h;
    out_h[b * E + e] = h;

    float ev = bp[e];
    const float* wp = Wp + (size_t)e * PL;
    for (int k = 0; k < PL; ++k) ev = fmaf(wp[k], sp[k], ev);
    enc[b * E + e] = ev;
}

// ---------------------------------------------------------------------------
// Score: p[b,m] = sum_e mem_hop[b,m,e]*q[b,e]   (mem recomputed from L2-resident table)
// optionally p_resto[b,m] = sum_e mem*q*enc. grid=B*M, block=E
// ---------------------------------------------------------------------------
__global__ void score_kernel(const float* __restrict__ table,   // tables + hop*NW*E
                             const int*   __restrict__ ctx,
                             const float* __restrict__ q,
                             const float* __restrict__ enc,
                             float* __restrict__ p_out,
                             float* __restrict__ presto_out,
                             int doResto) {
    const int bm = blockIdx.x;
    const int b  = bm >> 10;
    const int e  = threadIdx.x;
    const int* cp = ctx + (size_t)bm * F;
    float acc = 0.f;
#pragma unroll
    for (int f = 0; f < F; ++f) {
        const int idx = cp[f];
        if (idx) acc += table[(size_t)idx * E + e];
    }
    const float qe = q[b * E + e];
    float v1 = acc * qe;
    float v2 = doResto ? v1 * enc[b * E + e] : 0.f;

    const int wid = e >> 5, ln = e & 31;
    for (int off = 16; off; off >>= 1) {
        v1 += __shfl_down(v1, off, 32);
        v2 += __shfl_down(v2, off, 32);
    }
    __shared__ float r1[8], r2[8];
    if (ln == 0) { r1[wid] = v1; r2[wid] = v2; }
    __syncthreads();
    if (e == 0) {
        float s1 = 0.f, s2 = 0.f;
#pragma unroll
        for (int i = 0; i < 8; ++i) { s1 += r1[i]; s2 += r2[i]; }
        p_out[bm] = s1;
        if (doResto) presto_out[bm] = s2;
    }
}

// ---------------------------------------------------------------------------
// Softmax over M=1024 per batch row. grid=B, block=256 (4 elems/thread)
// ---------------------------------------------------------------------------
__global__ void softmax_kernel(const float* __restrict__ p, float* __restrict__ attn) {
    const int b = blockIdx.x, t = threadIdx.x;
    const float* row = p + b * M;
    float v[4];
#pragma unroll
    for (int i = 0; i < 4; ++i) v[i] = row[t + i * 256];
    float mx = fmaxf(fmaxf(v[0], v[1]), fmaxf(v[2], v[3]));
    const int wid = t >> 5, ln = t & 31;
    __shared__ float red[8];
    __shared__ float bc;
    for (int off = 16; off; off >>= 1) mx = fmaxf(mx, __shfl_down(mx, off, 32));
    if (ln == 0) red[wid] = mx;
    __syncthreads();
    if (t == 0) {
        float m2 = red[0];
#pragma unroll
        for (int i = 1; i < 8; ++i) m2 = fmaxf(m2, red[i]);
        bc = m2;
    }
    __syncthreads();
    mx = bc;
    float s = 0.f;
#pragma unroll
    for (int i = 0; i < 4; ++i) { v[i] = __expf(v[i] - mx); s += v[i]; }
    __syncthreads();
    for (int off = 16; off; off >>= 1) s += __shfl_down(s, off, 32);
    if (ln == 0) red[wid] = s;
    __syncthreads();
    if (t == 0) {
        float s2 = 0.f;
#pragma unroll
        for (int i = 0; i < 8; ++i) s2 += red[i];
        bc = 1.f / s2;
    }
    __syncthreads();
    const float inv = bc;
    float* arow = attn + b * M;
#pragma unroll
    for (int i = 0; i < 4; ++i) arow[t + i * 256] = v[i] * inv;
}

// ---------------------------------------------------------------------------
// o[b,e] = sum_m attn[b,m] * mem_{hop+1}[b,m,e];  q += o.
// For hop 0 also emit bf16 A-panel [h | o] (64x512) for the WMMA vocab GEMM.
// grid=B, block=E
// ---------------------------------------------------------------------------
__global__ void o_kernel(const float* __restrict__ table_next,
                         const int*   __restrict__ ctx,
                         const float* __restrict__ attn,
                         float* __restrict__ q,
                         unsigned short* __restrict__ Abf,
                         int writeA) {
    const int b = blockIdx.x, e = threadIdx.x;
    __shared__ int   sidx[M * F];
    __shared__ float sattn[M];
    for (int i = e; i < M * F; i += E) sidx[i] = ctx[(size_t)b * M * F + i];
    for (int i = e; i < M; i += E)     sattn[i] = attn[b * M + i];
    __syncthreads();
    float acc = 0.f;
    for (int m = 0; m < M; ++m) {
        const float a = sattn[m];
        float s = 0.f;
        const int base = m * F;
#pragma unroll
        for (int f = 0; f < F; ++f) {
            const int idx = sidx[base + f];
            if (idx) s += table_next[(size_t)idx * E + e];
        }
        acc = fmaf(a, s, acc);
    }
    const float q0 = q[b * E + e];
    if (writeA) {
        Abf[b * 2 * E + e]     = f2bf(q0);    // h half
        Abf[b * 2 * E + E + e] = f2bf(acc);   // o half
    }
    q[b * E + e] = q0 + acc;
}

// ---------------------------------------------------------------------------
// p_vocab = [64,512]bf16 @ Wv.T (bf16-converted) + bv, f32 accum via
// v_wmma_f32_16x16x32_bf16. One wave = one 16x16 output tile; the K loop is
// fully unrolled (16 WMMAs) so global_load_b128s pipeline across WMMAs.
// grid = 1000 blocks * 256 threads = 8000 waves = 4 row-tiles * 2000 col-tiles.
// ---------------------------------------------------------------------------
__global__ void pvocab_wmma_kernel(const unsigned short* __restrict__ Abf, // [64][512] bf16
                                   const float* __restrict__ Wv,            // [NW][512]
                                   const float* __restrict__ bv,
                                   float* __restrict__ out_pv) {            // [64][NW]
    const int gtid   = blockIdx.x * blockDim.x + threadIdx.x;
    const int wave   = gtid >> 5;
    const int lane   = threadIdx.x & 31;
    const int rowTile = wave & 3;        // 4 tiles of 16 batch rows
    const int colTile = wave >> 2;       // 2000 tiles of 16 vocab cols
    const int half = lane >> 4;
    const int ln   = lane & 15;
    const int m = rowTile * 16 + ln;     // A row (M) for this lane
    const int n = colTile * 16 + ln;     // Wv row (N) for this lane

    const unsigned short* arow = Abf + (size_t)m * (2 * E);
    const float*          wrow = Wv  + (size_t)n * (2 * E);

    v8f c = {};
#pragma unroll
    for (int k0 = 0; k0 < 2 * E; k0 += 32) {
        // A (16-bit 16x32 layout): a[i]   = A[m][k0 + half*8 + i]        i=0..7
        //                          a[8+i] = A[m][k0 + 16 + half*8 + i]
        const int abase = k0 + half * 8;
        const v8us alo = *(const v8us*)(arow + abase);        // 16B vector load
        const v8us ahi = *(const v8us*)(arow + abase + 16);   // 16B vector load
        v16us au;
#pragma unroll
        for (int j = 0; j < 8; ++j) { au[j] = alo[j]; au[8 + j] = ahi[j]; }
        const v16bf av = __builtin_bit_cast(v16bf, au);

        // B (16-bit 32x16 layout): b[i] = Wv[n][k0 + half*8 + i], b[8+i] = Wv[n][k0 + 16 + half*8 + i]
        const int bbase = k0 + half * 8;
        const v4f w0 = *(const v4f*)(wrow + bbase);           // 4 x 16B vector loads
        const v4f w1 = *(const v4f*)(wrow + bbase + 4);
        const v4f w2 = *(const v4f*)(wrow + bbase + 16);
        const v4f w3 = *(const v4f*)(wrow + bbase + 20);
        v16us bu;
#pragma unroll
        for (int j = 0; j < 4; ++j) {
            bu[j]      = f2bf(w0[j]);
            bu[4 + j]  = f2bf(w1[j]);
            bu[8 + j]  = f2bf(w2[j]);
            bu[12 + j] = f2bf(w3[j]);
        }
        const v16bf bvv = __builtin_bit_cast(v16bf, bu);

        c = __builtin_amdgcn_wmma_f32_16x16x32_bf16(false, av, false, bvv,
                                                    (short)0, c, false, false);
    }
    const float bias = bv[n];
    // D layout: lane -> N=ln ; VGPR vg -> M = vg + half*8
    float* outp = out_pv + (size_t)(rowTile * 16) * NW + (size_t)colTile * 16 + ln;
#pragma unroll
    for (int vg = 0; vg < 8; ++vg) {
        const int row = vg + half * 8;
        outp[(size_t)row * NW] = c[vg] + bias;
    }
}

// ---------------------------------------------------------------------------
extern "C" void kernel_launch(void* const* d_in, const int* in_sizes, int n_in,
                              void* d_out, int out_size, void* d_ws, size_t ws_size,
                              hipStream_t stream) {
    const float* tables = (const float*)d_in[0];
    const float* Wih    = (const float*)d_in[1];
    const float* Whh    = (const float*)d_in[2];
    const float* bih    = (const float*)d_in[3];
    const float* bhh    = (const float*)d_in[4];
    const float* Wp     = (const float*)d_in[5];
    const float* bp     = (const float*)d_in[6];
    const float* Wv     = (const float*)d_in[7];
    const float* bv     = (const float*)d_in[8];
    const int*   ctx    = (const int*)  d_in[9];
    const int*   y_     = (const int*)  d_in[10];
    const float* h_     = (const float*)d_in[11];
    const float* prof   = (const float*)d_in[12];

    float* out = (float*)d_out;
    float* out_p      = out;                         // [B,M]
    float* out_pvocab = out + B * M;                 // [B,NW]
    float* out_presto = out + B * M + B * NW;        // [B,M]
    float* out_h      = out + 2 * B * M + B * NW;    // [1,B,E]

    float* ws   = (float*)d_ws;
    float* q    = ws;                  // B*E
    float* enc  = ws + B * E;          // B*E
    float* praw = ws + 2 * B * E;      // B*M
    float* attn = praw + B * M;        // B*M
    unsigned short* Abf = (unsigned short*)(attn + B * M);  // B * 2E bf16

    const size_t TSZ = (size_t)NW * E;
    const float* tab0 = tables;
    const float* tab1 = tables + TSZ;
    const float* tab2 = tables + 2 * TSZ;

    // h = GRU(embed(tables[1], y), h_);  enc = prof @ Wp.T + bp ; q = h
    gru_enc_kernel<<<B, E, 0, stream>>>(tables, Wih, Whh, bih, bhh, Wp, bp,
                                        y_, h_, prof, q, enc, out_h);
    // hop 0
    score_kernel<<<B * M, E, 0, stream>>>(tab0, ctx, q, enc, praw, nullptr, 0);
    softmax_kernel<<<B, 256, 0, stream>>>(praw, attn);
    o_kernel<<<B, E, 0, stream>>>(tab1, ctx, attn, q, Abf, 1);   // q=q1, Abf=[h|o0]
    // p_vocab (independent of hops 1-2) — WMMA GEMM
    pvocab_wmma_kernel<<<(4 * (NW / 16)) / 8, 256, 0, stream>>>(Abf, Wv, bv, out_pvocab);
    // hop 1 (score also produces p_resto with q1 * enc)
    score_kernel<<<B * M, E, 0, stream>>>(tab1, ctx, q, enc, praw, out_presto, 1);
    softmax_kernel<<<B, 256, 0, stream>>>(praw, attn);
    o_kernel<<<B, E, 0, stream>>>(tab2, ctx, attn, q, nullptr, 0);  // q=q2
    // hop 2: only raw scores are live (returned p); softmax/o are dead code
    score_kernel<<<B * M, E, 0, stream>>>(tab2, ctx, q, enc, out_p, nullptr, 0);
}
